// BLAv2_30305289241010
// MI455X (gfx1250) — matrix operand order repaired and checked by reading, not compiled, not run on hardware
//
#include <hip/hip_runtime.h>
#include <hip/hip_bf16.h>
#include <math.h>

// ============================================================================
// MI455X (gfx1250) implementation of the 3-scale local-attention pyramid.
// All 1x1 convs -> bf16 WMMA GEMMs (v_wmma_f32_16x16x32_bf16, f32 accum).
// Each wave computes a 64x16 output strip (4 WMMA tiles) so the streamed
// X-fragment is reused 4x per load. Attention / LayerNorm stay in f32 VALU.
// ============================================================================

typedef __attribute__((ext_vector_type(16))) __bf16 v16bf;
typedef __attribute__((ext_vector_type(8)))  float  v8f;

#define HD 32  // head dim = FS[i]/HEADS[i] = 32 for every map
#define MT 4   // M sub-tiles per wave (64 output rows)

// ----------------------------------------------------------------------------
// GEMM: Y[b] (MxN) = W (MxK) @ X[b] (KxN) + bias, optional ReLU / +=.
// W row-major (K contiguous), X row-major (N contiguous).
// Requires M%64==0, N%128==0, K%32==0 (true for every conv in this net).
// Block: 256 threads = 8 waves tiling N; each wave owns 64x16 of Y.
// Fragment layouts per CDNA5 ISA (16-bit A 16x32, B 32x16, C/D 8xVGPR).
// ----------------------------------------------------------------------------
__global__ __launch_bounds__(256)
void gemm_bf16_wmma(const float* __restrict__ Wm, const float* __restrict__ Xm,
                    const float* __restrict__ bias, float* __restrict__ Ym,
                    int M, int N, int K, int relu, int accum)
{
  const int lane = threadIdx.x & 31;
  const int wave = threadIdx.x >> 5;
  const int half = lane >> 4;   // 0: lanes 0-15, 1: lanes 16-31
  const int l16  = lane & 15;
  const int b    = blockIdx.z;

  const float* X = Xm + (size_t)b * K * N;
  float*       Y = Ym + (size_t)b * M * N;
  const int m0 = blockIdx.y * (16 * MT);
  const int n0 = blockIdx.x * 128 + wave * 16;

  v8f acc[MT];
#pragma unroll
  for (int t = 0; t < MT; ++t) acc[t] = (v8f){};

  // B: col n0+l16; half 0 covers K 0..15, half 1 covers K 16..31
  const float* bcol = X + (size_t)(half * 16) * N + (n0 + l16);
  // A: row m0+l16 (+16 per sub-tile); half 0: K 0..15, half 1: K 16..31
  const float* arow0 = Wm + (size_t)(m0 + l16) * K + half * 16;

  for (int k0 = 0; k0 < K; k0 += 32) {
    v16bf bfr;
#pragma unroll
    for (int e = 0; e < 16; ++e) bfr[e] = (__bf16)bcol[(size_t)e * N];
    if (k0 + 32 < K)  // cover next k-step's strided X loads (global_prefetch_b8)
      __builtin_prefetch(bcol + (size_t)32 * N, 0, 1);
#pragma unroll
    for (int t = 0; t < MT; ++t) {
      const float* arow = arow0 + (size_t)(t * 16) * K + k0;
      v16bf af;
#pragma unroll
      for (int e = 0; e < 8; ++e)  af[e]     = (__bf16)arow[e];
#pragma unroll
      for (int e = 0; e < 8; ++e)  af[8 + e] = (__bf16)arow[8 + e];
      acc[t] = __builtin_amdgcn_wmma_f32_16x16x32_bf16(
          /*neg_a=*/false, af, /*neg_b=*/false, bfr,
          /*c_mod=*/(short)0, acc[t], /*reuse_a=*/false, /*reuse_b=*/false);
    }
    bcol += (size_t)32 * N;
  }

  // D layout: VGPR r -> M = m0 + t*16 + half*8 + r, N = n0 + l16
#pragma unroll
  for (int t = 0; t < MT; ++t) {
#pragma unroll
    for (int r = 0; r < 8; ++r) {
      int m = m0 + t * 16 + half * 8 + r;
      float vv = acc[t][r] + bias[m];
      if (relu) vv = fmaxf(vv, 0.f);
      float* dst = &Y[(size_t)m * N + (n0 + l16)];
      if (accum) *dst += vv; else *dst = vv;
    }
  }
}

// ----------------------------------------------------------------------------
// Local 3x3-window attention. One thread per (b, head, h, w). hd == 32.
// logits_j = s^2 * (q . k_j) + s * (q . pos_j)   (pos term present even when
// the window position is zero-padded OOB, matching the reference exactly).
// mode 0: self  (window center = (h, w) in k-map of same size)
// mode 1: td    (window center = (h>>1, w>>1) in coarse k-map; up2 semantics)
// mode 2: bu    (window positions = (2h+di, 2w+dj) in fine k-map; stride-2 unfold)
// ----------------------------------------------------------------------------
__global__ __launch_bounds__(256)
void local_attn_kernel(const float* __restrict__ q, unsigned long long qBS,
                       const float* __restrict__ k, const float* __restrict__ v,
                       unsigned long long kvBS,
                       const float* __restrict__ pos,
                       float* __restrict__ out, unsigned long long oBS,
                       int Bt, int heads, int Hq, int Wq, int Hk, int Wk,
                       float s, int mode)
{
  int idx = blockIdx.x * blockDim.x + threadIdx.x;
  int total = Bt * heads * Hq * Wq;
  if (idx >= total) return;
  int w = idx % Wq; int t = idx / Wq;
  int h = t % Hq;   t /= Hq;
  int n = t % heads; int b = t / heads;

  const size_t HWq = (size_t)Hq * Wq, HWk = (size_t)Hk * Wk;
  const float* qp = q + (size_t)b * qBS + (size_t)(n * HD) * HWq + (size_t)h * Wq + w;
  const float* kp = k + (size_t)b * kvBS + (size_t)(n * HD) * HWk;
  const float* vp = v + (size_t)b * kvBS + (size_t)(n * HD) * HWk;
  const float* pp = pos + (size_t)(n * HD) * 9;

  float qreg[HD];
#pragma unroll
  for (int d = 0; d < HD; ++d) qreg[d] = qp[(size_t)d * HWq];

  int bh, bw;
  if      (mode == 1) { bh = h >> 1; bw = w >> 1; }
  else if (mode == 2) { bh = h << 1; bw = w << 1; }
  else                { bh = h;      bw = w;      }

  float logits[9];
  float mx = -3.4e38f;
#pragma unroll
  for (int j = 0; j < 9; ++j) {
    int kh = bh + (j / 3) - 1, kw = bw + (j % 3) - 1;
    bool inb = (kh >= 0) && (kh < Hk) && (kw >= 0) && (kw < Wk);
    float dot = 0.f, dpos = 0.f;
    const float* kq = kp + (size_t)kh * Wk + kw;
#pragma unroll
    for (int d = 0; d < HD; ++d) {
      dpos += qreg[d] * pp[d * 9 + j];
      if (inb) dot += qreg[d] * kq[(size_t)d * HWk];
    }
    logits[j] = s * s * dot + s * dpos;
    mx = fmaxf(mx, logits[j]);
  }
  float ssum = 0.f;
#pragma unroll
  for (int j = 0; j < 9; ++j) { logits[j] = __expf(logits[j] - mx); ssum += logits[j]; }
  float inv = 1.f / ssum;

  float o[HD];
#pragma unroll
  for (int d = 0; d < HD; ++d) o[d] = 0.f;
#pragma unroll
  for (int j = 0; j < 9; ++j) {
    int kh = bh + (j / 3) - 1, kw = bw + (j % 3) - 1;
    if (kh < 0 || kh >= Hk || kw < 0 || kw >= Wk) continue;  // v zero-padded OOB
    float a = logits[j] * inv;
    const float* vq = vp + (size_t)kh * Wk + kw;
#pragma unroll
    for (int d = 0; d < HD; ++d) o[d] += a * vq[(size_t)d * HWk];
  }
  float* op = out + (size_t)b * oBS + (size_t)(n * HD) * HWq + (size_t)h * Wq + w;
#pragma unroll
  for (int d = 0; d < HD; ++d) op[(size_t)d * HWq] = o[d];
}

// ----------------------------------------------------------------------------
// y = LayerNorm_channels(x + delta) * g + beta. One thread per (b, pixel).
// Population variance, eps = 1e-5. Safe in-place (thread owns its column).
// ----------------------------------------------------------------------------
__global__ __launch_bounds__(256)
void ln_res_kernel(const float* __restrict__ x, const float* __restrict__ dl,
                   const float* __restrict__ g, const float* __restrict__ be,
                   float* __restrict__ y, int Bt, int C, int HW)
{
  int idx = blockIdx.x * blockDim.x + threadIdx.x;
  int total = Bt * HW;
  if (idx >= total) return;
  int p = idx % HW; int b = idx / HW;
  const float* xb = x  + (size_t)b * C * HW + p;
  const float* db = dl + (size_t)b * C * HW + p;
  float mean = 0.f;
  for (int c = 0; c < C; ++c) mean += xb[(size_t)c * HW] + db[(size_t)c * HW];
  mean /= (float)C;
  float var = 0.f;
  for (int c = 0; c < C; ++c) {
    float t = xb[(size_t)c * HW] + db[(size_t)c * HW] - mean;
    var += t * t;
  }
  float inv = rsqrtf(var / (float)C + 1e-5f);
  float* yb = y + (size_t)b * C * HW + p;
  for (int c = 0; c < C; ++c) {
    float t = xb[(size_t)c * HW] + db[(size_t)c * HW];
    yb[(size_t)c * HW] = (t - mean) * inv * g[c] + be[c];
  }
}

// ============================================================================
// Host orchestration
// ============================================================================
extern "C" void kernel_launch(void* const* d_in, const int* in_sizes, int n_in,
                              void* d_out, int out_size, void* d_ws, size_t ws_size,
                              hipStream_t stream)
{
  (void)in_sizes; (void)n_in; (void)out_size; (void)ws_size;
  const int Bt = 2, L = 2;
  const int FS_[3]    = {128, 256, 512};
  const int HEADS_[3] = {4, 8, 16};
  const int INF_[3]   = {512, 1024, 2048};
  const int HH[3]     = {128, 64, 32};
  const float SC = 0.59460355750136053f;  // 2^(-3/4) = 8^(-0.25), same for all maps

  auto F = [&](int i) -> const float* { return (const float*)d_in[i]; };
  // Leaf index bases in setup_inputs() dict / pytree order:
  // 0..2 x; 3 in_proj_w; 6 in_proj_b;
  // 9 self_qkv_w; 15 self_qkv_b; 21 self_out_w; 27 self_out_b; 33 self_pos;
  // 39 td_q_w; 43 td_q_b; 47 td_kv_w; 51 td_kv_b; 55 td_out_w; 59 td_out_b; 63 td_pos;
  // 67 bu_q_w; 71 bu_q_b; 75 bu_kv_w; 79 bu_kv_b; 83 bu_out_w; 87 bu_out_b; 91 bu_pos;
  // 95 attn_ln_g; 101 attn_ln_b; 107 ffn_in_w; 113 ffn_in_b; 119 ffn_out_w;
  // 125 ffn_out_b; 131 ffn_ln_g; 137 ffn_ln_b.  (total 143 leaves)

  // ---- workspace bump allocator (~160 MB total; all regions written before read)
  size_t off = 0;
  auto alloc = [&](size_t nf) { float* p = (float*)d_ws + off; off += nf; return p; };
  float* maps[3]; float* dl[3];
  for (int i = 0; i < 3; ++i) maps[i] = alloc((size_t)Bt * FS_[i] * HH[i] * HH[i]);
  for (int i = 0; i < 3; ++i) dl[i]   = alloc((size_t)Bt * FS_[i] * HH[i] * HH[i]);
  float* S1 = alloc((size_t)16777216);  // qkv / bu_kv / ffn-hidden (max 2*512*16384)
  float* S2 = alloc((size_t)4194304);   // td/bu q
  float* S3 = alloc((size_t)4194304);   // attn output / ffn y

  auto gemm = [&](const float* Wm, const float* Xm, const float* bias, float* Y,
                  int M, int N, int K, int relu, int accum) {
    dim3 grid(N / 128, M / 64, Bt);
    gemm_bf16_wmma<<<grid, dim3(256), 0, stream>>>(Wm, Xm, bias, Y, M, N, K, relu, accum);
  };
  auto attn = [&](const float* q, size_t qBS, const float* k, const float* v, size_t kvBS,
                  const float* pos, float* o, size_t oBS,
                  int heads, int Hq, int Hk, int mode) {
    int total = Bt * heads * Hq * Hq;
    local_attn_kernel<<<dim3((total + 255) / 256), dim3(256), 0, stream>>>(
        q, (unsigned long long)qBS, k, v, (unsigned long long)kvBS, pos,
        o, (unsigned long long)oBS, Bt, heads, Hq, Hq, Hk, Hk, SC, mode);
  };
  auto ln = [&](const float* x, const float* d, const float* g, const float* be,
                float* y, int C, int HW) {
    int total = Bt * HW;
    ln_res_kernel<<<dim3((total + 255) / 256), dim3(256), 0, stream>>>(
        x, d, g, be, y, Bt, C, HW);
  };

  // ---- input projections
  for (int i = 0; i < 3; ++i)
    gemm(F(3 + i), F(i), F(6 + i), maps[i], FS_[i], HH[i] * HH[i], INF_[i], 0, 0);

  for (int l = 0; l < L; ++l) {
    // -- self attention (3x3 local window on same map)
    for (int i = 0; i < 3; ++i) {
      int f = FS_[i], HWi = HH[i] * HH[i];
      gemm(F(9 + l * 3 + i), maps[i], F(15 + l * 3 + i), S1, 3 * f, HWi, f, 0, 0);
      attn(S1, (size_t)3 * f * HWi,
           S1 + (size_t)f * HWi, S1 + (size_t)2 * f * HWi, (size_t)3 * f * HWi,
           F(33 + l * 3 + i), S3, (size_t)f * HWi, HEADS_[i], HH[i], HH[i], 0);
      gemm(F(21 + l * 3 + i), S3, F(27 + l * 3 + i), dl[i], f, HWi, f, 0, 0);
    }
    // -- top-down: fine queries attend to 3x3 window around parent in coarse map
    for (int i = 0; i < 2; ++i) {
      int f = FS_[i], fc = FS_[i + 1];
      int HWq = HH[i] * HH[i], HWk = HH[i + 1] * HH[i + 1];
      gemm(F(39 + l * 2 + i), maps[i],     F(43 + l * 2 + i), S2, f,     HWq, f,  0, 0);
      gemm(F(47 + l * 2 + i), maps[i + 1], F(51 + l * 2 + i), S1, 2 * f, HWk, fc, 0, 0);
      attn(S2, (size_t)f * HWq, S1, S1 + (size_t)f * HWk, (size_t)2 * f * HWk,
           F(63 + l * 2 + i), S3, (size_t)f * HWq, HEADS_[i], HH[i], HH[i + 1], 1);
      gemm(F(55 + l * 2 + i), S3, F(59 + l * 2 + i), dl[i], f, HWq, f, 0, 1);
    }
    // -- bottom-up: coarse queries attend to stride-2 3x3 windows in fine map
    for (int i = 0; i < 2; ++i) {
      int fq = FS_[i + 1], ff = FS_[i];
      int HWq = HH[i + 1] * HH[i + 1], HWk = HH[i] * HH[i];
      gemm(F(67 + l * 2 + i), maps[i + 1], F(71 + l * 2 + i), S2, fq,     HWq, fq, 0, 0);
      gemm(F(75 + l * 2 + i), maps[i],     F(79 + l * 2 + i), S1, 2 * fq, HWk, ff, 0, 0);
      attn(S2, (size_t)fq * HWq, S1, S1 + (size_t)fq * HWk, (size_t)2 * fq * HWk,
           F(91 + l * 2 + i), S3, (size_t)fq * HWq, HEADS_[i + 1], HH[i + 1], HH[i], 2);
      gemm(F(83 + l * 2 + i), S3, F(87 + l * 2 + i), dl[i + 1], fq, HWq, fq, 0, 1);
    }
    // -- residual + LayerNorm (in place)
    for (int i = 0; i < 3; ++i)
      ln(maps[i], dl[i], F(95 + l * 3 + i), F(101 + l * 3 + i), maps[i],
         FS_[i], HH[i] * HH[i]);
    // -- FFN (4x expansion, ReLU) + residual LN (in place)
    for (int i = 0; i < 3; ++i) {
      int f = FS_[i], HWi = HH[i] * HH[i];
      gemm(F(107 + l * 3 + i), maps[i], F(113 + l * 3 + i), S1, 4 * f, HWi, f, 1, 0);
      gemm(F(119 + l * 3 + i), S1, F(125 + l * 3 + i), S3, f, HWi, 4 * f, 0, 0);
      ln(maps[i], S3, F(131 + l * 3 + i), F(137 + l * 3 + i), maps[i], f, HWi);
    }
  }

  // ---- outputs: tuple(map0, map1, map2) concatenated flat
  size_t o0 = (size_t)Bt * 128 * 128 * 128;
  size_t o1 = (size_t)Bt * 256 * 64 * 64;
  size_t o2 = (size_t)Bt * 512 * 32 * 32;
  hipMemcpyAsync(d_out, maps[0], o0 * sizeof(float), hipMemcpyDeviceToDevice, stream);
  hipMemcpyAsync((float*)d_out + o0, maps[1], o1 * sizeof(float), hipMemcpyDeviceToDevice, stream);
  hipMemcpyAsync((float*)d_out + o0 + o1, maps[2], o2 * sizeof(float), hipMemcpyDeviceToDevice, stream);
}
